// FASTKAGAT_6640019439799
// MI455X (gfx1250) — matrix operand order repaired and checked by reading, not compiled, not run on hardware
//
#include <hip/hip_runtime.h>
#include <hip/hip_bf16.h>
#include <math.h>

// ---------------------------------------------------------------------------
// FastKAN-GAT for MI455X (gfx1250). Dense GEMMs via v_wmma_f32_16x16x32_bf16
// with 4-way output-tile register blocking and explicit double-buffered
// operand loads (so WMMAs are not serialized behind s_wait_loadcnt 0).
// Graph scatter/softmax via coalesced float atomics. wave32 throughout.
// ---------------------------------------------------------------------------

typedef __attribute__((ext_vector_type(16))) __bf16 v16bf;
typedef __attribute__((ext_vector_type(8)))  float  v8f;

#define HEADS 4
#define HID   64
#define HC    256   // HEADS*HID
#define NCLS  16

// ---------------- utility kernels ----------------

__global__ void fk_fill_u32(unsigned* __restrict__ p, unsigned v, size_t n) {
    size_t i = (size_t)blockIdx.x * blockDim.x + threadIdx.x;
    if (i < n) p[i] = v;
}

__global__ void fk_f32_to_bf16(const float* __restrict__ a, __bf16* __restrict__ o, size_t n) {
    size_t i = (size_t)blockIdx.x * blockDim.x + threadIdx.x;
    if (i < n) o[i] = (__bf16)a[i];
}

// ---------------- LayerNorm + RBF basis expansion ----------------
// One block per node; blockDim.x == D (128 or 256). Output: [N, D*4] bf16 row-major.
__global__ void fk_ln_rbf(const float* __restrict__ x, const float* __restrict__ gamma,
                          const float* __restrict__ beta, __bf16* __restrict__ out, int D) {
    int n = blockIdx.x;
    int t = threadIdx.x;
    float v = x[(size_t)n * D + t];
    float s = v, s2 = v * v;
#pragma unroll
    for (int off = 16; off > 0; off >>= 1) {
        s  += __shfl_xor(s,  off);
        s2 += __shfl_xor(s2, off);
    }
    __shared__ float sh[16];
    __shared__ float mv[2];
    int nw  = blockDim.x >> 5;
    int wid = t >> 5;
    if ((t & 31) == 0) { sh[wid] = s; sh[8 + wid] = s2; }
    __syncthreads();
    if (t == 0) {
        float S = 0.f, S2 = 0.f;
        for (int i = 0; i < nw; i++) { S += sh[i]; S2 += sh[8 + i]; }
        float mean = S / (float)D;
        float var  = S2 / (float)D - mean * mean;
        mv[0] = mean;
        mv[1] = rsqrtf(var + 1e-5f);
    }
    __syncthreads();
    float hn = (v - mv[0]) * mv[1] * gamma[t] + beta[t];
    const float grid4[4] = {-2.0f, -0.66666667f, 0.66666667f, 2.0f};
    size_t base = (size_t)n * D * 4 + (size_t)t * 4;
#pragma unroll
    for (int g = 0; g < 4; g++) {
        float u = (hn - grid4[g]) * 0.75f;   // 1/DENOM = 3/4
        out[base + g] = (__bf16)__expf(-u * u);
    }
}

// ---------------- WMMA GEMM: C[nRows,O] = A[nRows,K] * B[O,K]^T ----------------
// A,B bf16 row-major; C f32. 8 waves per block; each wave owns one 16-row tile
// and NTO consecutive 16-col output tiles. Two-deep software pipeline on the
// K dimension (K must be a multiple of 64).
template <int NTO>
__global__ void __launch_bounds__(256)
fk_wmma_gemm(const __bf16* __restrict__ A, const __bf16* __restrict__ B,
             float* __restrict__ C, int nRows, int K, int O) {
    int lane  = threadIdx.x & 31;
    int wave  = threadIdx.x >> 5;
    int mtile = blockIdx.x * 8 + wave;
    if (mtile * 16 >= nRows) return;
    int obase = blockIdx.y * (NTO * 16);

    int arow = mtile * 16 + (lane & 15);
    // A layout: lanes 0-15 row M, K {k0..k0+7}/{k0+16..+23}; lanes 16-31 same rows at +8/+24.
    const __bf16* aPtr = A + (size_t)arow * K + ((lane & 16) ? 8 : 0);
    // B layout: lane = column N; lanes 0-15 hold K {k0..k0+7}/{k0+16..+23}; lanes 16-31 at +8/+24.
    const __bf16* bPtr[NTO];
#pragma unroll
    for (int t = 0; t < NTO; t++) {
        int bcol = obase + t * 16 + (lane & 15);
        bPtr[t] = B + (size_t)bcol * K + ((lane & 16) ? 8 : 0);
    }

    union U { v16bf v; uint4 q[2]; };
    v8f acc[NTO];
#pragma unroll
    for (int t = 0; t < NTO; t++) acc[t] = (v8f){0.f, 0.f, 0.f, 0.f, 0.f, 0.f, 0.f, 0.f};

    U a0, a1;
    U b0[NTO], b1[NTO];

    auto loadA = [&](U& a, int k) {
        a.q[0] = *(const uint4*)(aPtr + k);
        a.q[1] = *(const uint4*)(aPtr + k + 16);
    };
    auto loadB = [&](U* b, int k) {
#pragma unroll
        for (int t = 0; t < NTO; t++) {
            b[t].q[0] = *(const uint4*)(bPtr[t] + k);
            b[t].q[1] = *(const uint4*)(bPtr[t] + k + 16);
        }
    };
    auto mul = [&](U& a, U* b) {
#pragma unroll
        for (int t = 0; t < NTO; t++) {
            acc[t] = __builtin_amdgcn_wmma_f32_16x16x32_bf16(
                false, a.v, false, b[t].v, (short)0, acc[t], false, false);
        }
    };

    // two-deep software pipeline over K (K % 64 == 0)
    loadA(a0, 0);
    loadB(b0, 0);
    for (int k0 = 0; k0 < K; k0 += 64) {
        loadA(a1, k0 + 32);             // stage 1 loads issue before stage 0 compute
        loadB(b1, k0 + 32);
        if (k0 + 64 < K) __builtin_prefetch(aPtr + k0 + 64, 0, 3);  // global_prefetch_b8
        mul(a0, b0);
        if (k0 + 64 < K) {
            loadA(a0, k0 + 64);         // stage 0 loads issue before stage 1 compute
            loadB(b0, k0 + 64);
        }
        mul(a1, b1);
    }

    // C/D layout: VGPR r: lanes 0-15 -> M=r, lanes 16-31 -> M=8+r; N = lane&15.
    int rbase = (lane & 16) ? 8 : 0;
#pragma unroll
    for (int t = 0; t < NTO; t++) {
        int col = obase + t * 16 + (lane & 15);
#pragma unroll
        for (int r = 0; r < 8; r++) {
            int row = mtile * 16 + rbase + r;
            C[(size_t)row * O + col] = acc[t][r];
        }
    }
}

// ---------------- attention coefficients ----------------
__global__ void fk_alpha(const float* __restrict__ h, const float* __restrict__ att_s,
                         const float* __restrict__ att_d, float* __restrict__ oas,
                         float* __restrict__ oad, int N) {
    int idx = blockIdx.x * blockDim.x + threadIdx.x;
    if (idx >= N * HEADS) return;
    int n = idx >> 2, hd = idx & 3;
    const float* hp = h + (size_t)n * HC + hd * HID;
    const float* sp = att_s + hd * HID;
    const float* dp = att_d + hd * HID;
    float ss = 0.f, sd = 0.f;
#pragma unroll 8
    for (int c = 0; c < HID; c++) { float v = hp[c]; ss += v * sp[c]; sd += v * dp[c]; }
    oas[idx] = ss;
    oad[idx] = sd;
}

// ---- order-preserving float<->uint encoding for atomic segment-max ----
__device__ __forceinline__ unsigned fk_mono(float f) {
    unsigned u = __float_as_uint(f);
    return (u & 0x80000000u) ? ~u : (u | 0x80000000u);
}
__device__ __forceinline__ float fk_mono_dec(unsigned u) {
    unsigned b = (u & 0x80000000u) ? (u & 0x7fffffffu) : ~u;
    return __uint_as_float(b);
}

__device__ __forceinline__ void fk_edge_ends(const int* ei, int idx, int E, int& s, int& d) {
    if (idx < E) { s = ei[idx]; d = ei[E + idx]; }
    else         { s = idx - E; d = s; }          // self-loops appended
}

__global__ void fk_edge_max(const int* __restrict__ ei, const float* __restrict__ as_,
                            const float* __restrict__ ad_, unsigned* __restrict__ emax,
                            int E, int N) {
    int idx = blockIdx.x * blockDim.x + threadIdx.x;
    if (idx >= E + N) return;
    int s, d; fk_edge_ends(ei, idx, E, s, d);
#pragma unroll
    for (int h = 0; h < HEADS; h++) {
        float e = as_[s * HEADS + h] + ad_[d * HEADS + h];
        e = (e > 0.f) ? e : 0.2f * e;             // leaky_relu 0.2
        atomicMax(&emax[d * HEADS + h], fk_mono(e));
    }
}

__global__ void fk_edge_exp(const int* __restrict__ ei, const float* __restrict__ as_,
                            const float* __restrict__ ad_, const unsigned* __restrict__ emax,
                            float* __restrict__ ee, float* __restrict__ denom, int E, int N) {
    int idx = blockIdx.x * blockDim.x + threadIdx.x;
    if (idx >= E + N) return;
    int s, d; fk_edge_ends(ei, idx, E, s, d);
#pragma unroll
    for (int h = 0; h < HEADS; h++) {
        float e = as_[s * HEADS + h] + ad_[d * HEADS + h];
        e = (e > 0.f) ? e : 0.2f * e;
        float v = __expf(e - fk_mono_dec(emax[d * HEADS + h]));
        ee[(size_t)idx * HEADS + h] = v;
        atomicAdd(&denom[d * HEADS + h], v);
    }
}

// One block per edge, 256 threads cover all HC channels -> coalesced atomics.
__global__ void fk_edge_aggr(const int* __restrict__ ei, const float* __restrict__ h,
                             const float* __restrict__ ee, const float* __restrict__ denom,
                             float* __restrict__ out, int E, int N) {
    int b = blockIdx.x;
    int t = threadIdx.x;                 // channel 0..255
    int s, d; fk_edge_ends(ei, b, E, s, d);
    int hd = t >> 6;
    float w = ee[(size_t)b * HEADS + hd] / denom[d * HEADS + hd];
    atomicAdd(&out[(size_t)d * HC + t], h[(size_t)s * HC + t] * w);
}

__global__ void fk_bias_silu(float* __restrict__ x, const float* __restrict__ bias, size_t n) {
    size_t i = (size_t)blockIdx.x * blockDim.x + threadIdx.x;
    if (i >= n) return;
    float v = x[i] + bias[i & (HC - 1)];
    x[i] = v / (1.f + __expf(-v));
}

__global__ void fk_pool(const float* __restrict__ x, const int* __restrict__ batch,
                        float* __restrict__ pooled, size_t n) {
    size_t i = (size_t)blockIdx.x * blockDim.x + threadIdx.x;
    if (i >= n) return;
    int node = (int)(i >> 8);
    int c    = (int)(i & (HC - 1));
    atomicAdd(&pooled[(size_t)batch[node] * HC + c], x[i]);
}

__global__ void fk_logsoftmax(const float* __restrict__ logits, float* __restrict__ out,
                              int nGraphs) {
    int g = threadIdx.x;
    if (g >= nGraphs) return;
    float v[NCLS];
    float m = -1e30f;
#pragma unroll
    for (int c = 0; c < NCLS; c++) { v[c] = logits[g * NCLS + c]; m = fmaxf(m, v[c]); }
    float s = 0.f;
#pragma unroll
    for (int c = 0; c < NCLS; c++) s += __expf(v[c] - m);
    float ls = __logf(s) + m;
#pragma unroll
    for (int c = 0; c < NCLS; c++) out[g * NCLS + c] = v[c] - ls;
}

// ---------------------------------------------------------------------------

extern "C" void kernel_launch(void* const* d_in, const int* in_sizes, int n_in,
                              void* d_out, int out_size, void* d_ws, size_t ws_size,
                              hipStream_t stream) {
    const float* x      = (const float*)d_in[0];
    const int*   ei     = (const int*)  d_in[1];
    const int*   batch  = (const int*)  d_in[2];
    const float* ln_g0  = (const float*)d_in[3];
    const float* ln_b0  = (const float*)d_in[4];
    const float* W0     = (const float*)d_in[5];
    const float* att_s0 = (const float*)d_in[6];
    const float* att_d0 = (const float*)d_in[7];
    const float* bias0  = (const float*)d_in[8];
    const float* ln_g1  = (const float*)d_in[9];
    const float* ln_b1  = (const float*)d_in[10];
    const float* W1     = (const float*)d_in[11];
    const float* att_s1 = (const float*)d_in[12];
    const float* att_d1 = (const float*)d_in[13];
    const float* bias1  = (const float*)d_in[14];
    const float* ln_gr  = (const float*)d_in[15];
    const float* ln_br  = (const float*)d_in[16];
    const float* Wr     = (const float*)d_in[17];

    const int F_IN = 128;
    const int N = in_sizes[0] / F_IN;     // 50000
    const int E = in_sizes[1] / 2;        // 800000
    const int NG = out_size / NCLS;       // 64 graphs
    const int K0 = F_IN * 4;              // 512
    const int K1 = HC * 4;                // 1024

    // ---- workspace carve-out ----
    char* wp = (char*)d_ws;
    auto alloc = [&](size_t bytes) -> void* {
        void* p = wp;
        wp += (bytes + 255) & ~(size_t)255;
        return p;
    };
    __bf16*  W0bf   = (__bf16*) alloc((size_t)HC * K0 * 2);
    __bf16*  W1bf   = (__bf16*) alloc((size_t)HC * K1 * 2);
    __bf16*  Wrbf   = (__bf16*) alloc((size_t)NCLS * K1 * 2);
    __bf16*  basis  = (__bf16*) alloc((size_t)N * K1 * 2);        // reused for K0 & K1
    float*   hbuf   = (float*)  alloc((size_t)N * HC * 4);
    float*   aggbuf = (float*)  alloc((size_t)N * HC * 4);
    float*   as_    = (float*)  alloc((size_t)N * HEADS * 4);
    float*   ad_    = (float*)  alloc((size_t)N * HEADS * 4);
    unsigned* emax  = (unsigned*)alloc((size_t)N * HEADS * 4);
    float*   denom  = (float*)  alloc((size_t)N * HEADS * 4);
    float*   ee     = (float*)  alloc((size_t)(E + N) * HEADS * 4);
    float*   pooled = (float*)  alloc((size_t)NG * HC * 4);
    __bf16*  basisr = (__bf16*) alloc((size_t)NG * K1 * 2);
    float*   logits = (float*)  alloc((size_t)NG * NCLS * 4);

    const unsigned NEG_INF_MONO = 0x007FFFFFu;  // fk_mono(-inf)
    const int nTiles = (N + 15) / 16;
    dim3 gemmBlk(256);

    auto cvt = [&](const float* src, __bf16* dst, size_t n) {
        fk_f32_to_bf16<<<(unsigned)((n + 255) / 256), 256, 0, stream>>>(src, dst, n);
    };
    cvt(W0, W0bf, (size_t)HC * K0);
    cvt(W1, W1bf, (size_t)HC * K1);
    cvt(Wr, Wrbf, (size_t)NCLS * K1);

    auto gat_layer = [&](const float* xin, int D, int K, const float* lng, const float* lnb,
                         const __bf16* Wbf, const float* atts, const float* attd,
                         const float* bias) {
        // 1) LN + RBF basis
        fk_ln_rbf<<<N, D, 0, stream>>>(xin, lng, lnb, basis, D);
        // 2) h = basis @ W^T  (WMMA bf16->f32, 4 o-tiles per wave, double-buffered)
        fk_wmma_gemm<4><<<dim3((nTiles + 7) / 8, HC / 64), gemmBlk, 0, stream>>>(
            basis, Wbf, hbuf, N, K, HC);
        // 3) attention coefficients
        fk_alpha<<<(N * HEADS + 255) / 256, 256, 0, stream>>>(hbuf, atts, attd, as_, ad_, N);
        // 4) init reductions
        fk_fill_u32<<<(N * HEADS + 255) / 256, 256, 0, stream>>>(emax, NEG_INF_MONO,
                                                                 (size_t)N * HEADS);
        fk_fill_u32<<<(N * HEADS + 255) / 256, 256, 0, stream>>>((unsigned*)denom, 0u,
                                                                 (size_t)N * HEADS);
        fk_fill_u32<<<(unsigned)(((size_t)N * HC + 255) / 256), 256, 0, stream>>>(
            (unsigned*)aggbuf, 0u, (size_t)N * HC);
        // 5) segment softmax over dst
        int nEdge = E + N;
        fk_edge_max<<<(nEdge + 255) / 256, 256, 0, stream>>>(ei, as_, ad_, emax, E, N);
        fk_edge_exp<<<(nEdge + 255) / 256, 256, 0, stream>>>(ei, as_, ad_, emax, ee, denom, E, N);
        // 6) weighted aggregation (coalesced per-edge block)
        fk_edge_aggr<<<nEdge, HC, 0, stream>>>(ei, hbuf, ee, denom, aggbuf, E, N);
        // 7) bias + SiLU (in place)
        fk_bias_silu<<<(unsigned)(((size_t)N * HC + 255) / 256), 256, 0, stream>>>(
            aggbuf, bias, (size_t)N * HC);
    };

    // layer 0: F_IN=128 -> HC;  layer 1: HC -> HC (input = aggbuf from layer 0)
    gat_layer(x,      F_IN, K0, ln_g0, ln_b0, W0bf, att_s0, att_d0, bias0);
    gat_layer(aggbuf, HC,   K1, ln_g1, ln_b1, W1bf, att_s1, att_d1, bias1);

    // global_add_pool
    fk_fill_u32<<<(NG * HC + 255) / 256, 256, 0, stream>>>((unsigned*)pooled, 0u,
                                                           (size_t)NG * HC);
    fk_pool<<<(unsigned)(((size_t)N * HC + 255) / 256), 256, 0, stream>>>(
        aggbuf, batch, pooled, (size_t)N * HC);

    // readout FastKAN: LN+RBF -> WMMA GEMM [NG x NCLS] -> log_softmax
    fk_ln_rbf<<<NG, HC, 0, stream>>>(pooled, ln_gr, ln_br, basisr, HC);
    fk_wmma_gemm<1><<<dim3(((NG + 15) / 16 + 7) / 8, NCLS / 16), gemmBlk, 0, stream>>>(
        basisr, Wrbf, logits, NG, K1, NCLS);
    fk_logsoftmax<<<1, 64, 0, stream>>>(logits, (float*)d_out, NG);

    (void)n_in; (void)ws_size;
}
